// Head_43250320671481
// MI455X (gfx1250) — matrix operand order repaired and checked by reading, not compiled, not run on hardware
//
#include <hip/hip_runtime.h>

typedef __attribute__((ext_vector_type(16))) _Float16 v16h;
typedef __attribute__((ext_vector_type(8)))  float    v8f;

// Payload/pointer types for the gfx1250 async global->LDS copy builtin:
// parameter type is 'vector_size(16) int' pointers in AS1 (global) / AS3 (LDS).
typedef int v4i_vs __attribute__((vector_size(16)));
typedef __attribute__((address_space(1))) v4i_vs* as1_v4i;
typedef __attribute__((address_space(3))) v4i_vs* as3_v4i;

#define BB 4
#define TT 2048
#define CC 1024
#define HH 64

// ---------------------------------------------------------------------------
// Async 16-byte global->LDS copy (gfx1250 GLOBAL_LOAD_ASYNC_TO_LDS_B128,
// tracked with ASYNCcnt).  Builtin when available, inline asm otherwise.
// ---------------------------------------------------------------------------
__device__ __forceinline__ void async_cp16(const void* g, void* l) {
#if __has_builtin(__builtin_amdgcn_global_load_async_to_lds_b128)
    __builtin_amdgcn_global_load_async_to_lds_b128(
        (as1_v4i)(__attribute__((address_space(1))) void*)(void*)g,
        (as3_v4i)(__attribute__((address_space(3))) void*)(void*)l,
        0, 0);
#else
    // GV mode: vdst = LDS byte offset (addr[31:0]), vaddr = 64-bit global.
    asm volatile("global_load_async_to_lds_b128 %0, %1, off"
                 :: "v"((unsigned)(size_t)l),
                    "v"((unsigned long long)(size_t)g)
                 : "memory");
#endif
}

// ---------------------------------------------------------------------------
// Kernel 0: one-shot fp32 -> f16 conversion of Wq|Wk|Wv into workspace.
// ---------------------------------------------------------------------------
__global__ void __launch_bounds__(256) convert_w(
    const float* __restrict__ Wq, const float* __restrict__ Wk,
    const float* __restrict__ Wv, _Float16* __restrict__ Wh)
{
    const int per = CC * HH;
    int i = blockIdx.x * blockDim.x + threadIdx.x;
    if (i < per)            Wh[i] = (_Float16)Wq[i];
    else if (i < 2 * per)   Wh[i] = (_Float16)Wk[i - per];
    else if (i < 3 * per)   Wh[i] = (_Float16)Wv[i - 2 * per];
}

// ---------------------------------------------------------------------------
// Kernel 1: QKV projection.  grid = B*(T/16) blocks of 96 threads (3 waves).
// Wave 0 -> Q, wave 1 -> K, wave 2 -> V.  Each wave produces a 16x64 f16 tile
// via v_wmma_f32_16x16x32_f16, K-loop over C=1024 in steps of 32.
// Weights are pre-converted f16 (no cvt VALU in the inner loop).
// ---------------------------------------------------------------------------
__global__ void __launch_bounds__(96) proj_qkv(
    const float* __restrict__ x,
    const _Float16* __restrict__ Wh,
    const float* __restrict__ bq, const float* __restrict__ bk,
    const float* __restrict__ bv,
    _Float16* __restrict__ Qh, _Float16* __restrict__ Kh,
    _Float16* __restrict__ Vh)
{
    const int wave = threadIdx.x >> 5;
    const int lane = threadIdx.x & 31;
    const int tilesPerB = TT / 16;
    const int b  = blockIdx.x / tilesPerB;
    const int t0 = (blockIdx.x % tilesPerB) * 16;

    const _Float16* W   = Wh + (size_t)wave * CC * HH;
    const float*    bia = (wave == 0) ? bq : (wave == 1) ? bk : bv;
    _Float16*       out = (wave == 0) ? Qh : (wave == 1) ? Kh : Vh;

    // A operand, this code's contiguous k order: lane holds row M = lane%16,
    // K base = (lane/16)*16; VGPR i: K = base+2i, base+2i+1.
    const int m    = lane & 15;
    const int k0   = (lane >> 4) * 16;
    // B operand, same contiguous k order as A: lane holds col N = lane%16,
    // K base = (lane/16)*16, VGPR j packs K = base+2j, base+2j+1.
    const int n    = lane & 15;
    const int kb   = (lane >> 4) * 16;
    const int half = lane >> 4;

    const float* xrow = x + ((size_t)b * TT + t0 + m) * CC;

    v8f acc[4] = {};

    for (int kc = 0; kc < CC; kc += 32) {
        v16h a;
        #pragma unroll
        for (int i = 0; i < 8; ++i) {
            int kk = kc + k0 + 2 * i;
            a[2 * i]     = (_Float16)xrow[kk];
            a[2 * i + 1] = (_Float16)xrow[kk + 1];
        }
        #pragma unroll
        for (int nt = 0; nt < 4; ++nt) {
            v16h bm;
            #pragma unroll
            for (int j = 0; j < 8; ++j) {
                int kk = kc + kb + 2 * j;
                bm[2 * j]     = W[(size_t)kk * HH + nt * 16 + n];
                bm[2 * j + 1] = W[(size_t)(kk + 1) * HH + nt * 16 + n];
            }
            acc[nt] = __builtin_amdgcn_wmma_f32_16x16x32_f16(
                false, a, false, bm, (short)0, acc[nt], false, false);
        }
    }

    // f32 C/D 16x16 layout: lane half owns rows half*8 + i, col = lane%16.
    #pragma unroll
    for (int nt = 0; nt < 4; ++nt) {
        float bb_ = bia[nt * 16 + n];
        #pragma unroll
        for (int i = 0; i < 8; ++i) {
            int row = t0 + i + half * 8;
            out[((size_t)b * TT + row) * HH + nt * 16 + n] =
                (_Float16)(acc[nt][i] + bb_);
        }
    }
}

// ---------------------------------------------------------------------------
// Kernel 2: fused causal attention (flash style).
// grid = B*(T/64) blocks of 128 threads (4 waves).  All 4 waves share one
// 64-query tile and iterate a uniform 32-key loop; K/V tiles are staged into
// double-buffered LDS with GLOBAL_LOAD_ASYNC_TO_LDS_B128, prefetching tile
// i+1 while computing tile i (s_wait_asynccnt 4 exploits in-order async
// completion).  Each wave: S = Q K^T (chained K=32 WMMAs), online softmax
// (fully-masked tiles are numeric no-ops: alpha=1, p=0), P restage through
// per-wave LDS to A-layout, O += P V.
// ---------------------------------------------------------------------------
__global__ void __launch_bounds__(128) attn_head(
    const _Float16* __restrict__ Qh, const _Float16* __restrict__ Kh,
    const _Float16* __restrict__ Vh, float* __restrict__ out)
{
    __shared__ _Float16 kbuf[2][32 * 64];    // 2 x 4 KB: 32 keys x 64 dims
    __shared__ _Float16 vbuf[2][32 * 64];    // 2 x 4 KB
    __shared__ _Float16 pstage[4][16 * 32];  // per-wave P restage

    const int tid  = threadIdx.x;
    const int wave = tid >> 5;
    const int lane = tid & 31;
    const int tilesPerB = TT / 64;
    const int b     = blockIdx.x / tilesPerB;
    const int q0blk = (blockIdx.x % tilesPerB) * 64;
    const int q0    = q0blk + wave * 16;

    const int m    = lane & 15;
    const int k0   = (lane >> 4) * 16;
    const int half = lane >> 4;
    const int n    = lane & 15;
    const int kb   = half * 16;

    // Stage one 32-key K/V tile pair into LDS buffer `bi` (4 async B128 each).
    auto stage = [&](int s0, int bi) {
        const char* kg = (const char*)(Kh + ((size_t)b * TT + s0) * HH);
        const char* vg = (const char*)(Vh + ((size_t)b * TT + s0) * HH);
        char* kl = (char*)kbuf[bi];
        char* vl = (char*)vbuf[bi];
        const int off = tid * 16;
        async_cp16(kg + off,        kl + off);
        async_cp16(kg + off + 2048, kl + off + 2048);
        async_cp16(vg + off,        vl + off);
        async_cp16(vg + off + 2048, vl + off + 2048);
    };

    // Load Q tile as two A-operands covering head dims 0..31 / 32..63.
    const _Float16* qrow = Qh + ((size_t)b * TT + q0 + m) * HH;
    v16h aq0, aq1;
    #pragma unroll
    for (int i = 0; i < 8; ++i) {
        int kk = k0 + 2 * i;
        aq0[2 * i]     = qrow[kk];
        aq0[2 * i + 1] = qrow[kk + 1];
        aq1[2 * i]     = qrow[32 + kk];
        aq1[2 * i + 1] = qrow[32 + kk + 1];
    }

    float mrow[8], lrow[8];
    #pragma unroll
    for (int i = 0; i < 8; ++i) { mrow[i] = -1e30f; lrow[i] = 0.0f; }
    v8f o[4] = {};

    const int smax = q0blk + 63;   // uniform across the block
    stage(0, 0);                   // prologue: tile 0 in flight

    int it = 0;
    for (int s0 = 0; s0 <= smax; s0 += 32, ++it) {
        const int  bi      = it & 1;
        const bool hasNext = (s0 + 32) <= smax;

        // Prefetch next tile into the other buffer, then wait for this one.
        if (hasNext) {
            stage(s0 + 32, bi ^ 1);
            asm volatile("s_wait_asynccnt 4" ::: "memory");
        } else {
            asm volatile("s_wait_asynccnt 0" ::: "memory");
        }
        __syncthreads();

        const _Float16* kt = kbuf[bi];
        const _Float16* vt = vbuf[bi];

        // ---- S = Q K^T over keys [s0, s0+32): two 16-key column tiles ----
        v8f s[2];
        #pragma unroll
        for (int jt = 0; jt < 2; ++jt) {
            // B-operand: element (kdim, col) = K[key = s0+jt*16+col][kdim]
            const _Float16* krow = kt + (jt * 16 + n) * HH;
            v16h b0, b1;
            #pragma unroll
            for (int j = 0; j < 8; ++j) {
                int kk = kb + 2 * j;
                b0[2 * j]     = krow[kk];
                b0[2 * j + 1] = krow[kk + 1];
                b1[2 * j]     = krow[32 + kk];
                b1[2 * j + 1] = krow[32 + kk + 1];
            }
            v8f t = {};
            t = __builtin_amdgcn_wmma_f32_16x16x32_f16(
                false, aq0, false, b0, (short)0, t, false, false);
            t = __builtin_amdgcn_wmma_f32_16x16x32_f16(
                false, aq1, false, b1, (short)0, t, false, false);
            s[jt] = t;
        }

        // ---- causal mask + online softmax (per-row over 16-lane halves) ---
        float cand[8];
        #pragma unroll
        for (int i = 0; i < 8; ++i) {
            int qr = q0 + i + half * 8;
            float v0 = ((s0 + n)      > qr) ? -1e30f : s[0][i];
            float v1 = ((s0 + 16 + n) > qr) ? -1e30f : s[1][i];
            s[0][i] = v0; s[1][i] = v1;
            cand[i] = fmaxf(v0, v1);
        }
        #pragma unroll
        for (int off = 8; off >= 1; off >>= 1) {
            #pragma unroll
            for (int i = 0; i < 8; ++i)
                cand[i] = fmaxf(cand[i], __shfl_xor(cand[i], off, 16));
        }
        float alpha[8], rs[8];
        #pragma unroll
        for (int i = 0; i < 8; ++i) {
            float nm = fmaxf(mrow[i], cand[i]);
            alpha[i] = __expf(mrow[i] - nm);
            mrow[i]  = nm;
            float p0 = __expf(s[0][i] - nm);
            float p1 = __expf(s[1][i] - nm);
            s[0][i] = p0; s[1][i] = p1;
            rs[i] = p0 + p1;
        }
        #pragma unroll
        for (int off = 8; off >= 1; off >>= 1) {
            #pragma unroll
            for (int i = 0; i < 8; ++i)
                rs[i] += __shfl_xor(rs[i], off, 16);
        }
        #pragma unroll
        for (int i = 0; i < 8; ++i)
            lrow[i] = lrow[i] * alpha[i] + rs[i];
        #pragma unroll
        for (int nt = 0; nt < 4; ++nt) {
            #pragma unroll
            for (int i = 0; i < 8; ++i)
                o[nt][i] *= alpha[i];
        }

        // ---- stage P (D-layout) -> LDS -> reload in A-layout ----
        _Float16* pst = pstage[wave];
        #pragma unroll
        for (int i = 0; i < 8; ++i) {
            int row = i + half * 8;
            pst[row * 32 + n]      = (_Float16)s[0][i];
            pst[row * 32 + 16 + n] = (_Float16)s[1][i];
        }
        asm volatile("s_wait_dscnt 0" ::: "memory");
        v16h ap;
        #pragma unroll
        for (int i = 0; i < 8; ++i) {
            int kk = k0 + 2 * i;
            ap[2 * i]     = pst[m * 32 + kk];
            ap[2 * i + 1] = pst[m * 32 + kk + 1];
        }

        // ---- O += P @ V  (B-operand: rows = keys, cols = head dim) ----
        #pragma unroll
        for (int nt = 0; nt < 4; ++nt) {
            v16h bvm;
            #pragma unroll
            for (int j = 0; j < 8; ++j) {
                int kr = kb + 2 * j;
                bvm[2 * j]     = vt[kr * HH + nt * 16 + n];
                bvm[2 * j + 1] = vt[(kr + 1) * HH + nt * 16 + n];
            }
            o[nt] = __builtin_amdgcn_wmma_f32_16x16x32_f16(
                false, ap, false, bvm, (short)0, o[nt], false, false);
        }
        __syncthreads();   // all waves done reading before next overwrite
    }

    // ---- normalize and store fp32 output ----
    #pragma unroll
    for (int nt = 0; nt < 4; ++nt) {
        #pragma unroll
        for (int i = 0; i < 8; ++i) {
            int row = q0 + i + half * 8;
            out[((size_t)b * TT + row) * HH + nt * 16 + n] = o[nt][i] / lrow[i];
        }
    }
}

extern "C" void kernel_launch(void* const* d_in, const int* in_sizes, int n_in,
                              void* d_out, int out_size, void* d_ws, size_t ws_size,
                              hipStream_t stream) {
    const float* x  = (const float*)d_in[0];
    const float* Wq = (const float*)d_in[1];
    const float* bq = (const float*)d_in[2];
    const float* Wk = (const float*)d_in[3];
    const float* bk = (const float*)d_in[4];
    const float* Wv = (const float*)d_in[5];
    const float* bv = (const float*)d_in[6];
    float* out = (float*)d_out;

    const size_t qkv_elems = (size_t)BB * TT * HH;
    _Float16* Qh = (_Float16*)d_ws;
    _Float16* Kh = Qh + qkv_elems;
    _Float16* Vh = Kh + qkv_elems;
    _Float16* Wh = Vh + qkv_elems;          // 3 * CC * HH f16

    const int wtot = 3 * CC * HH;
    convert_w<<<dim3((wtot + 255) / 256), dim3(256), 0, stream>>>(Wq, Wk, Wv, Wh);
    proj_qkv<<<dim3(BB * (TT / 16)), dim3(96), 0, stream>>>(
        x, Wh, bq, bk, bv, Qh, Kh, Vh);
    attn_head<<<dim3(BB * (TT / 64)), dim3(128), 0, stream>>>(
        Qh, Kh, Vh, out);
}
